// GraphTransformerEncoder_65463891526136
// MI455X (gfx1250) — compile-verified, matrix operand order checked by reading, not run on hardware
//
#include <hip/hip_runtime.h>
#include <hip/hip_bf16.h>

// ---------------- problem constants (match reference) ----------------
#define NN   20000
#define EE   160000
#define INF_ 128
#define HID  96
#define NH   6
#define NG   512
#define NCLS 128
#define HD   576           // NH * HID
#define ARC_S 30.0f
#define INV_SQRT_HID 0.10206207261596575f  // 1/sqrt(96)

typedef __bf16 v8bf  __attribute__((ext_vector_type(8)));
typedef __bf16 v16bf __attribute__((ext_vector_type(16)));
typedef float  v8f   __attribute__((ext_vector_type(8)));

#define LDS_AS __attribute__((address_space(3)))

// true LDS byte offset of a pointer into __shared__ memory
__device__ __forceinline__ unsigned lds_byte_off(const void* p) {
    return (unsigned)(unsigned long long)(LDS_AS const char*)(const char*)p;
}

// 16x32 bf16 fragment: two contiguous 16B halves (K 0..7 / 16..23 per lane group)
__device__ __forceinline__ v16bf load_frag(const __bf16* p) {
    v16bf f;
    v8bf lo = *(const v8bf*)p;
    v8bf hi = *(const v8bf*)(p + 16);
#pragma unroll
    for (int j = 0; j < 8; ++j) { f[j] = lo[j]; f[8 + j] = hi[j]; }
    return f;
}

// ---------------- small utility kernels ----------------
__global__ void cast_f32_bf16_kernel(const float* __restrict__ src,
                                     __bf16* __restrict__ dst, long long n) {
    long long i = (long long)blockIdx.x * blockDim.x + threadIdx.x;
    if (i < n) dst[i] = (__bf16)src[i];
}

// W: K x F row-major f32  ->  WT: F x K row-major bf16
__global__ void transpose_cast_bf16_kernel(const float* __restrict__ W,
                                           __bf16* __restrict__ WT, int K, int F) {
    long long i = (long long)blockIdx.x * blockDim.x + threadIdx.x;
    if (i >= (long long)K * F) return;
    int k = (int)(i / F), f = (int)(i % F);
    WT[(long long)f * K + k] = (__bf16)W[i];
}

// ---------------- WMMA bf16 GEMM: C[M,F] = A[M,K] @ B[K,F] + bias ----------------
// A:  M x K row-major bf16 (streamed from global / L2).
// BT: F x K row-major bf16 (transposed weights). The block's N-panel of BT
//     (N_BLK x K, contiguous) is staged once into LDS with async b128 copies.
// Block = 256 threads = 8 waves; wave w computes a 32-row x (16*NT)-col macro
// tile (MT=2 m-tiles reuse each LDS B fragment twice). KC templated -> full unroll.
template <int NT, int KC>
__global__ __launch_bounds__(256)
void gemm_bf16_wmma_kernel(const __bf16* __restrict__ A,
                           const __bf16* __restrict__ BT,
                           const float* __restrict__ bias,
                           float* __restrict__ C,
                           int M, int F) {
    extern __shared__ __bf16 smemB[];          // N_BLK * KC bf16
    const int NBLK    = 16 * NT;
    const int nChunks = F / NBLK;
    const int bm = blockIdx.x / nChunks;
    const int n0 = (blockIdx.x % nChunks) * NBLK;

    // ---- stage B panel: global -> LDS, asynchronous (ASYNCcnt) ----
    {
        const char*    gsrc  = (const char*)(BT + (long long)n0 * KC);
        const unsigned bytes = (unsigned)(NBLK * KC * 2);
        const unsigned lbase = lds_byte_off(smemB);
        for (unsigned ofs = (unsigned)threadIdx.x * 16u; ofs < bytes; ofs += 256u * 16u) {
            const char* g = gsrc + ofs;
            unsigned    l = lbase + ofs;
            asm volatile("global_load_async_to_lds_b128 %0, %1, off"
                         :: "v"(l), "v"(g) : "memory");
        }
        asm volatile("s_wait_asynccnt 0x0" ::: "memory");
    }
    __syncthreads();

    const int lane = threadIdx.x & 31;
    const int wave = threadIdx.x >> 5;
    const int pair = bm * 8 + wave;            // 32-row macro-tile index
    const int mPairs = M >> 5;                 // M/32 (20000 -> 625, exact)
    if (pair >= mPairs) return;                // idle waves (no more barriers below)

    const int r    = lane & 15;                // row (A) / col (B,C)
    const int kgrp = (lane >> 4) * 8;          // lanes 16-31 take K offset +8 / +24

    v8f acc[2][NT] = {};
    const __bf16* arow0 = A + ((long long)pair * 32 + r) * KC;
    const __bf16* arow1 = arow0 + (long long)16 * KC;

#pragma unroll
    for (int kc = 0; kc < KC; kc += 32) {
        const int kb = kc + kgrp;
        __builtin_prefetch(arow0 + kb + 32, 0, 0);   // global_prefetch_b8
        v16bf a0 = load_frag(arow0 + kb);
        v16bf a1 = load_frag(arow1 + kb);
#pragma unroll
        for (int i = 0; i < NT; ++i) {
            v16bf bfrag = load_frag(smemB + (i * 16 + r) * KC + kb);  // ds_load_b128
            acc[0][i] = __builtin_amdgcn_wmma_f32_16x16x32_bf16(
                            false, a0, false, bfrag, (short)0, acc[0][i], false, false);
            acc[1][i] = __builtin_amdgcn_wmma_f32_16x16x32_bf16(
                            false, a1, false, bfrag, (short)0, acc[1][i], false, false);
        }
    }

    // C/D layout: VGPR v -> M = v (+8 for hi lanes); lane&15 -> N
#pragma unroll
    for (int mt = 0; mt < 2; ++mt) {
        const int rowBase = pair * 32 + mt * 16 + (lane >> 4) * 8;
#pragma unroll
        for (int i = 0; i < NT; ++i) {
            const int col = n0 + i * 16 + r;
            const float bv = bias[col];
#pragma unroll
            for (int v = 0; v < 8; ++v)
                C[(long long)(rowBase + v) * F + col] = acc[mt][i][v] + bv;
        }
    }
}

// ---------------- edge attention ----------------
__device__ __forceinline__ unsigned enc_f32(float f) {
    unsigned u = __float_as_uint(f);
    return (u & 0x80000000u) ? ~u : (u | 0x80000000u);
}
__device__ __forceinline__ float dec_f32(unsigned u) {
    return (u & 0x80000000u) ? __uint_as_float(u & 0x7FFFFFFFu) : __uint_as_float(~u);
}

__global__ void edge_alpha_kernel(const float* __restrict__ q, const float* __restrict__ k,
                                  const long long* __restrict__ ei,
                                  float* __restrict__ alpha, unsigned* __restrict__ mmax) {
    long long i = (long long)blockIdx.x * blockDim.x + threadIdx.x;
    if (i >= (long long)EE * NH) return;
    int e = (int)(i / NH), h = (int)(i % NH);
    long long s = ei[e], d = ei[EE + e];
    const float4* qp = (const float4*)(q + d * (long long)HD + h * HID);
    const float4* kp = (const float4*)(k + s * (long long)HD + h * HID);
    float acc = 0.f;
#pragma unroll
    for (int j = 0; j < HID / 4; ++j) {
        float4 a = qp[j], b = kp[j];
        acc += a.x * b.x + a.y * b.y + a.z * b.z + a.w * b.w;
    }
    acc *= INV_SQRT_HID;
    alpha[i] = acc;
    atomicMax(mmax + d * NH + h, enc_f32(acc));
}

__global__ void edge_exp_kernel(const long long* __restrict__ ei,
                                float* __restrict__ alpha,
                                const unsigned* __restrict__ mmax,
                                float* __restrict__ ssum) {
    long long i = (long long)blockIdx.x * blockDim.x + threadIdx.x;
    if (i >= (long long)EE * NH) return;
    int e = (int)(i / NH), h = (int)(i % NH);
    long long d = ei[EE + e];
    float a = expf(alpha[i] - dec_f32(mmax[d * NH + h]));
    alpha[i] = a;
    atomicAdd(ssum + d * NH + h, a);
}

__global__ void edge_norm_kernel(const long long* __restrict__ ei,
                                 float* __restrict__ alpha,
                                 const float* __restrict__ ssum) {
    long long i = (long long)blockIdx.x * blockDim.x + threadIdx.x;
    if (i >= (long long)EE * NH) return;
    int e = (int)(i / NH), h = (int)(i % NH);
    long long d = ei[EE + e];
    alpha[i] = alpha[i] / ssum[d * NH + h];
}

__global__ void edge_aggregate_kernel(const long long* __restrict__ ei,
                                      const float* __restrict__ alpha,
                                      const float* __restrict__ v,
                                      float* __restrict__ agg) {
    long long i = (long long)blockIdx.x * blockDim.x + threadIdx.x;
    if (i >= (long long)EE * HD) return;
    int e = (int)(i / HD), c = (int)(i % HD);
    int h = c / HID;
    long long s = ei[e], d = ei[EE + e];
    atomicAdd(agg + d * (long long)HD + c, alpha[(long long)e * NH + h] * v[s * (long long)HD + c]);
}

__global__ void node_finalize_kernel(const float* __restrict__ agg,
                                     const float* __restrict__ skip,
                                     float* __restrict__ hf,
                                     __bf16* __restrict__ hbf) {
    long long i = (long long)blockIdx.x * blockDim.x + threadIdx.x;
    if (i >= (long long)NN * HID) return;
    int n = (int)(i / HID), dch = (int)(i % HID);
    float acc = 0.f;
#pragma unroll
    for (int h = 0; h < NH; ++h) acc += agg[(long long)n * HD + h * HID + dch];
    float val = fmaxf(acc * (1.0f / NH) + skip[i], 0.0f);
    hf[i] = val;
    hbf[i] = (__bf16)val;
}

// ---------------- pooling + head ----------------
__global__ void pool_kernel(const float* __restrict__ h,
                            const long long* __restrict__ batch,
                            float* __restrict__ gsum, float* __restrict__ gcnt) {
    long long i = (long long)blockIdx.x * blockDim.x + threadIdx.x;
    if (i >= (long long)NN * HID) return;
    int n = (int)(i / HID), dch = (int)(i % HID);
    long long b = batch[n];
    atomicAdd(gsum + b * HID + dch, h[i]);
    if (dch == 0) atomicAdd(gcnt + b, 1.0f);
}

__global__ void head_kernel(const float* __restrict__ gsum, const float* __restrict__ gcnt,
                            const float* __restrict__ fc1w, const float* __restrict__ fc1b,
                            const float* __restrict__ fc2w, const float* __restrict__ fc2b,
                            const float* __restrict__ arcw, float* __restrict__ out) {
    __shared__ float gv[HID], t1[HID], t2[HID];
    __shared__ float gnorm;
    int g = blockIdx.x, tid = threadIdx.x;
    float denom = fmaxf(gcnt[g], 1.0f);
    if (tid < HID) gv[tid] = gsum[g * HID + tid] / denom;
    __syncthreads();
    if (tid < HID) {
        float a = fc1b[tid];
        for (int k = 0; k < HID; ++k) a += gv[k] * fc1w[k * HID + tid];
        t1[tid] = fmaxf(a, 0.f);
    }
    __syncthreads();
    if (tid < HID) {
        float a = fc2b[tid];
        for (int k = 0; k < HID; ++k) a += t1[k] * fc2w[k * HID + tid];
        t2[tid] = fmaxf(a, 0.f);
    }
    __syncthreads();
    if (tid == 0) {
        float s = 0.f;
        for (int k = 0; k < HID; ++k) s += t2[k] * t2[k];
        gnorm = sqrtf(s) + 1e-12f;
    }
    __syncthreads();
    if (tid < NCLS) {
        const float* w = arcw + tid * HID;
        float dot = 0.f, wn = 0.f;
        for (int k = 0; k < HID; ++k) { dot += t2[k] * w[k]; wn += w[k] * w[k]; }
        out[g * NCLS + tid] = ARC_S * dot / (gnorm * (sqrtf(wn) + 1e-12f));
    }
}

// ---------------- host orchestration ----------------
static inline int blks(long long n, int t) { return (int)((n + t - 1) / t); }

extern "C" void kernel_launch(void* const* d_in, const int* in_sizes, int n_in,
                              void* d_out, int out_size, void* d_ws, size_t ws_size,
                              hipStream_t stream) {
    const float*     x       = (const float*)d_in[0];
    const long long* ei      = (const long long*)d_in[1];
    const long long* batch   = (const long long*)d_in[2];
    const float *Wq1 = (const float*)d_in[3],  *bq1 = (const float*)d_in[4];
    const float *Wk1 = (const float*)d_in[5],  *bk1 = (const float*)d_in[6];
    const float *Wv1 = (const float*)d_in[7],  *bv1 = (const float*)d_in[8];
    const float *Ws1 = (const float*)d_in[9],  *bs1 = (const float*)d_in[10];
    const float *Wq_r = (const float*)d_in[11], *bq_r = (const float*)d_in[12];
    const float *Wk_r = (const float*)d_in[13], *bk_r = (const float*)d_in[14];
    const float *Wv_r = (const float*)d_in[15], *bv_r = (const float*)d_in[16];
    const float *Ws_r = (const float*)d_in[17], *bs_r = (const float*)d_in[18];
    const float *fc1w = (const float*)d_in[19], *fc1b = (const float*)d_in[20];
    const float *fc2w = (const float*)d_in[21], *fc2b = (const float*)d_in[22];
    const float *arcw = (const float*)d_in[23];
    float* out = (float*)d_out;

    // workspace carve-up (all chunks 256B aligned)
    char* p = (char*)d_ws;
    size_t off = 0;
    auto take = [&](size_t bytes) { void* r = p + off; off = (off + bytes + 255) & ~(size_t)255; return r; };
    __bf16*   x_bf  = (__bf16*)take((size_t)NN * INF_ * 2);
    __bf16*   h_bf  = (__bf16*)take((size_t)NN * HID * 2);
    __bf16*   wq_bf = (__bf16*)take((size_t)INF_ * HD * 2);
    __bf16*   wk_bf = (__bf16*)take((size_t)INF_ * HD * 2);
    __bf16*   wv_bf = (__bf16*)take((size_t)INF_ * HD * 2);
    __bf16*   ws_bf = (__bf16*)take((size_t)INF_ * HID * 2);
    float*    qb    = (float*)take((size_t)NN * HD * 4);
    float*    kb    = (float*)take((size_t)NN * HD * 4);
    float*    vb    = (float*)take((size_t)NN * HD * 4);
    float*    skip  = (float*)take((size_t)NN * HID * 4);
    float*    alpha = (float*)take((size_t)EE * NH * 4);
    unsigned* mmax  = (unsigned*)take((size_t)NN * NH * 4);
    float*    ssum  = (float*)take((size_t)NN * NH * 4);
    float*    agg   = (float*)take((size_t)NN * HD * 4);
    float*    h_f   = (float*)take((size_t)NN * HID * 4);
    float*    gsum  = (float*)take((size_t)NG * HID * 4);
    float*    gcnt  = (float*)take((size_t)NG * 4);
    (void)ws_size; (void)in_sizes; (void)n_in; (void)out_size;

    // cast node features once
    cast_f32_bf16_kernel<<<blks((long long)NN * INF_, 256), 256, 0, stream>>>(x, x_bf, (long long)NN * INF_);

    const long long eh = (long long)EE * NH;
    const int mPairBlocks = ((NN / 32) + 7) / 8;   // 8 waves x 32 rows per block

    for (int layer = 0; layer < 5; ++layer) {
        const int K = (layer == 0) ? INF_ : HID;
        const __bf16* A = (layer == 0) ? x_bf : h_bf;
        const float *Wq, *bq, *Wk, *bk_, *Wv, *bv, *Ws, *bs;
        if (layer == 0) {
            Wq = Wq1; bq = bq1; Wk = Wk1; bk_ = bk1; Wv = Wv1; bv = bv1; Ws = Ws1; bs = bs1;
        } else {
            int i = layer - 1;
            Wq = Wq_r + (size_t)i * HID * HD;  bq  = bq_r + (size_t)i * HD;
            Wk = Wk_r + (size_t)i * HID * HD;  bk_ = bk_r + (size_t)i * HD;
            Wv = Wv_r + (size_t)i * HID * HD;  bv  = bv_r + (size_t)i * HD;
            Ws = Ws_r + (size_t)i * HID * HID; bs  = bs_r + (size_t)i * HID;
        }

        // weights -> bf16 transposed [F, K]
        transpose_cast_bf16_kernel<<<blks((long long)K * HD, 256), 256, 0, stream>>>(Wq, wq_bf, K, HD);
        transpose_cast_bf16_kernel<<<blks((long long)K * HD, 256), 256, 0, stream>>>(Wk, wk_bf, K, HD);
        transpose_cast_bf16_kernel<<<blks((long long)K * HD, 256), 256, 0, stream>>>(Wv, wv_bf, K, HD);
        transpose_cast_bf16_kernel<<<blks((long long)K * HID, 256), 256, 0, stream>>>(Ws, ws_bf, K, HID);

        // projections (WMMA, B-panel staged to LDS via async copies)
        const int g576 = mPairBlocks * (HD / 64);              // NT=4 -> N_BLK=64
        const int g96  = mPairBlocks * (HID / 32);             // NT=2 -> N_BLK=32
        if (layer == 0) {
            const size_t l576 = (size_t)64 * INF_ * 2, l96 = (size_t)32 * INF_ * 2;
            gemm_bf16_wmma_kernel<4, INF_><<<g576, 256, l576, stream>>>(A, wq_bf, bq,  qb, NN, HD);
            gemm_bf16_wmma_kernel<4, INF_><<<g576, 256, l576, stream>>>(A, wk_bf, bk_, kb, NN, HD);
            gemm_bf16_wmma_kernel<4, INF_><<<g576, 256, l576, stream>>>(A, wv_bf, bv,  vb, NN, HD);
            gemm_bf16_wmma_kernel<2, INF_><<<g96, 256, l96, stream>>>(A, ws_bf, bs, skip, NN, HID);
        } else {
            const size_t l576 = (size_t)64 * HID * 2, l96 = (size_t)32 * HID * 2;
            gemm_bf16_wmma_kernel<4, HID><<<g576, 256, l576, stream>>>(A, wq_bf, bq,  qb, NN, HD);
            gemm_bf16_wmma_kernel<4, HID><<<g576, 256, l576, stream>>>(A, wk_bf, bk_, kb, NN, HD);
            gemm_bf16_wmma_kernel<4, HID><<<g576, 256, l576, stream>>>(A, wv_bf, bv,  vb, NN, HD);
            gemm_bf16_wmma_kernel<2, HID><<<g96, 256, l96, stream>>>(A, ws_bf, bs, skip, NN, HID);
        }

        // segment softmax state
        hipMemsetAsync(mmax, 0, (size_t)NN * NH * 4, stream);
        hipMemsetAsync(ssum, 0, (size_t)NN * NH * 4, stream);
        hipMemsetAsync(agg, 0, (size_t)NN * HD * 4, stream);

        edge_alpha_kernel<<<blks(eh, 256), 256, 0, stream>>>(qb, kb, ei, alpha, mmax);
        edge_exp_kernel<<<blks(eh, 256), 256, 0, stream>>>(ei, alpha, mmax, ssum);
        edge_norm_kernel<<<blks(eh, 256), 256, 0, stream>>>(ei, alpha, ssum);
        edge_aggregate_kernel<<<blks((long long)EE * HD, 256), 256, 0, stream>>>(ei, alpha, vb, agg);
        node_finalize_kernel<<<blks((long long)NN * HID, 256), 256, 0, stream>>>(agg, skip, h_f, h_bf);
    }

    // global mean pool + MLP head + ArcFace logits
    hipMemsetAsync(gsum, 0, (size_t)NG * HID * 4, stream);
    hipMemsetAsync(gcnt, 0, (size_t)NG * 4, stream);
    pool_kernel<<<blks((long long)NN * HID, 256), 256, 0, stream>>>(h_f, batch, gsum, gcnt);
    head_kernel<<<NG, 128, 0, stream>>>(gsum, gcnt, fc1w, fc1b, fc2w, fc2b, arcw, out);
}
